// NonLocalBlock2d_80358838108601
// MI455X (gfx1250) — matrix operand order repaired and checked by reading, not compiled, not run on hardware
//
#include <hip/hip_runtime.h>
#include <hip/hip_bf16.h>

// ---------------------------------------------------------------------------
// NonLocalBlock2d for MI455X (gfx1250): bf16 WMMA flash-attention pipeline.
// B=4, C=256, CI=128, H=W=64, N=4096.
// ---------------------------------------------------------------------------

#define B_  4
#define C_  256
#define CI_ 128
#define N_  4096
#define BN_EPS 1e-5f

typedef __attribute__((ext_vector_type(16))) __bf16 v16bf;
typedef __attribute__((ext_vector_type(8)))  __bf16 v8bf;
typedef __attribute__((ext_vector_type(2)))  __bf16 v2bf;
typedef __attribute__((ext_vector_type(8)))  float  v8f;

// Packed fp32->bf16 conversion: hardware v_cvt_pk_bf16_f32 (probe-confirmed).
__device__ __forceinline__ v2bf cvtpk(float a, float b) {
#if __has_builtin(__builtin_amdgcn_cvt_pk_bf16_f32)
    return __builtin_amdgcn_cvt_pk_bf16_f32(a, b);
#else
    v2bf t; t[0] = (__bf16)a; t[1] = (__bf16)b; return t;
#endif
}

__device__ __forceinline__ __bf16 f2bf(float f) {
    return cvtpk(f, 0.f)[0];
}

__device__ __forceinline__ v8bf cvt8(float4 a, float4 b) {
    v2bf p0 = cvtpk(a.x, a.y), p1 = cvtpk(a.z, a.w);
    v2bf p2 = cvtpk(b.x, b.y), p3 = cvtpk(b.z, b.w);
    v8bf t;
    t[0]=p0[0]; t[1]=p0[1]; t[2]=p1[0]; t[3]=p1[1];
    t[4]=p2[0]; t[5]=p2[1]; t[6]=p3[0]; t[7]=p3[1];
    return t;
}

// Build an A-operand fragment from a row-major bf16 matrix (row stride = ld).
// lane L: row = L&15, two 8-elem chunks at K = k0+16h and K = k0+16h+8.
__device__ __forceinline__ v16bf load_afrag(const __bf16* base, int ld, int m, int h, int k0) {
    const __bf16* p = base + m * ld + k0 + 16 * h;      // 32-bit offsets
    v8bf lo = *(const v8bf*)(p);
    v8bf hi = *(const v8bf*)(p + 8);
    v16bf t;
#pragma unroll
    for (int i = 0; i < 8; ++i) { t[i] = lo[i]; t[8 + i] = hi[i]; }
    return t;
}

#define WMMA_BF16(a, b, c) \
    __builtin_amdgcn_wmma_f32_16x16x32_bf16(false, (a), false, (b), (short)0, (c), false, false)

// ---------------------------------------------------------------------------
// K0: one-shot weight conversion fp32 -> bf16 (theta_w, phi_w, g_w, wz_w).
// ---------------------------------------------------------------------------
__global__ __launch_bounds__(256) void wcvt_kernel(
    const float* __restrict__ tw, const float* __restrict__ pw,
    const float* __restrict__ gw, const float* __restrict__ wz,
    __bf16* __restrict__ twb, __bf16* __restrict__ pwb,
    __bf16* __restrict__ gwb, __bf16* __restrict__ wzb)
{
    const int i = blockIdx.x * 256 + threadIdx.x;   // 0..32767
    twb[i] = f2bf(tw[i]);
    pwb[i] = f2bf(pw[i]);
    gwb[i] = f2bf(gw[i]);
    wzb[i] = f2bf(wz[i]);
}

// ---------------------------------------------------------------------------
// K1: projections. out[n,o] = sum_c x[b,c,n] * W[o,c] + bias[o]
// theta, phiT stored (B,N,CI); gT stored (B,CI,N). One wave = 16 n-rows.
// ---------------------------------------------------------------------------
__global__ __launch_bounds__(128) void proj_kernel(
    const float* __restrict__ x,
    const __bf16* __restrict__ tw, const float* __restrict__ tb,
    const __bf16* __restrict__ pw, const float* __restrict__ pb,
    const __bf16* __restrict__ gw, const float* __restrict__ gb,
    __bf16* __restrict__ theta, __bf16* __restrict__ phiT, __bf16* __restrict__ gT)
{
    __shared__ __bf16 lds[4][C_][16];      // per-wave x tile (c-major), 32 KB
    const int wave = threadIdx.x >> 5;
    const int lane = threadIdx.x & 31;
    const int gw_id = blockIdx.x * 4 + wave;
    const int b  = gw_id / (N_ / 16);
    const int n0 = (gw_id % (N_ / 16)) * 16;
    const int m = lane & 15, h = lane >> 4;

    // stage x[b, :, n0:n0+16] into LDS as bf16
    const float* xb = x + b * (C_ * N_);
    for (int c = lane; c < C_; c += 32) {
        const float* src = xb + c * N_ + n0;
        float4 f0 = *(const float4*)(src + 0);
        float4 f1 = *(const float4*)(src + 4);
        float4 f2 = *(const float4*)(src + 8);
        float4 f3 = *(const float4*)(src + 12);
        *(v8bf*)&lds[wave][c][0] = cvt8(f0, f1);
        *(v8bf*)&lds[wave][c][8] = cvt8(f2, f3);
    }
    __syncthreads();

    // A fragments for all 8 k-steps (x^T rows): lane row m, gathered from LDS
    v16bf afrag[8];
#pragma unroll
    for (int k = 0; k < 8; ++k) {
        const int c0 = k * 32;
        v16bf a;
#pragma unroll
        for (int i = 0; i < 8; ++i) {
            a[i]     = lds[wave][c0 + 16 * h + i][m];
            a[8 + i] = lds[wave][c0 + 16 * h + 8 + i][m];
        }
        afrag[k] = a;
    }

#pragma unroll
    for (int wsel = 0; wsel < 3; ++wsel) {
        const __bf16* Wm = (wsel == 0) ? tw : ((wsel == 1) ? pw : gw);
        const float*  Bv = (wsel == 0) ? tb : ((wsel == 1) ? pb : gb);
#pragma unroll
        for (int t = 0; t < CI_ / 16; ++t) {            // 8 output tiles
            v8f acc = {0.f,0.f,0.f,0.f,0.f,0.f,0.f,0.f};
            const int co = t * 16 + m;                  // this lane's output col
            const __bf16* wrow = Wm + co * C_ + 16 * h;
            v16bf bf = *(const v16bf*)(wrow);           // software-pipelined k-loop
#pragma unroll
            for (int k = 0; k < 8; ++k) {
                v16bf bfn = bf;
                if (k < 7) bfn = *(const v16bf*)(wrow + (k + 1) * 32);
                acc = WMMA_BF16(afrag[k], bf, acc);
                bf = bfn;
            }
            const float bias = Bv[co];
            if (wsel == 2) {                            // gT: (B,CI,N)
                __bf16* dst = gT + (b * CI_ + co) * N_ + n0;
#pragma unroll
                for (int v = 0; v < 8; ++v) dst[v + 8 * h] = f2bf(acc[v] + bias);
            } else {                                    // theta/phiT: (B,N,CI)
                __bf16* dst = ((wsel == 0) ? theta : phiT) + (b * N_ + n0) * CI_ + co;
#pragma unroll
                for (int v = 0; v < 8; ++v) dst[(v + 8 * h) * CI_] = f2bf(acc[v] + bias);
            }
        }
    }
}

// ---------------------------------------------------------------------------
// K2: flash attention. One wave (32-thread block) owns 16 query rows,
// streams 64-wide KV blocks with online softmax. y stored (B,N,CI) bf16.
// ---------------------------------------------------------------------------
__global__ __launch_bounds__(32) void attn_kernel(
    const __bf16* __restrict__ theta, const __bf16* __restrict__ phiT,
    const __bf16* __restrict__ gT, __bf16* __restrict__ yout)
{
    __shared__ __bf16 plds[16][72];     // P transpose buffer, padded stride
    const int lane = threadIdx.x & 31;
    const int m = lane & 15, h = lane >> 4;
    const int b  = blockIdx.x / (N_ / 16);
    const int n0 = (blockIdx.x % (N_ / 16)) * 16;

    const __bf16* th = theta + (b * N_ + n0) * CI_;
    const __bf16* ph = phiT + b * (N_ * CI_);
    const __bf16* gt = gT + b * (CI_ * N_);

    v16bf afrag[4];
#pragma unroll
    for (int k = 0; k < 4; ++k) afrag[k] = load_afrag(th, CI_, m, h, k * 32);

    float mrow[8], lrow[8];
    v8f yacc[8];
#pragma unroll
    for (int v = 0; v < 8; ++v) { mrow[v] = -3.0e38f; lrow[v] = 0.f; }
#pragma unroll
    for (int t = 0; t < 8; ++t) yacc[t] = (v8f){0.f,0.f,0.f,0.f,0.f,0.f,0.f,0.f};

    for (int j0 = 0; j0 < N_; j0 += 64) {
        if (j0 + 64 < N_) {   // uniform-condition prefetch of next KV block
            __builtin_prefetch((const void*)(ph + (j0 + 64 + m) * CI_), 0, 1);
            __builtin_prefetch((const void*)(gt + m * N_ + j0 + 64), 0, 1);
        }
        // ---- S = theta @ phi  (16 x 64), pipelined over 16 (jt,k) steps ----
        v8f s[4];
#pragma unroll
        for (int jt = 0; jt < 4; ++jt) s[jt] = (v8f){0.f,0.f,0.f,0.f,0.f,0.f,0.f,0.f};
        {
            const int base = (j0 + m) * CI_ + 16 * h;   // + jt*16*CI_ + k*32
            v16bf bf = *(const v16bf*)(ph + base);
#pragma unroll
            for (int i = 0; i < 16; ++i) {
                const int jt = i >> 2, k = i & 3;
                v16bf bfn = bf;
                if (i < 15) {
                    const int jn = (i + 1) >> 2, kn = (i + 1) & 3;
                    bfn = *(const v16bf*)(ph + base + jn * (16 * CI_) + kn * 32);
                }
                s[jt] = WMMA_BF16(afrag[k], bf, s[jt]);
                bf = bfn;
            }
        }
        // ---- online softmax: new row max (reduce over 4 tiles + 16 lanes) ----
        float nm[8];
#pragma unroll
        for (int v = 0; v < 8; ++v) {
            float t = fmaxf(fmaxf(s[0][v], s[1][v]), fmaxf(s[2][v], s[3][v]));
#pragma unroll
            for (int off = 1; off < 16; off <<= 1) t = fmaxf(t, __shfl_xor(t, off, 32));
            nm[v] = fmaxf(mrow[v], t);
        }
        // rescale running accumulators
#pragma unroll
        for (int v = 0; v < 8; ++v) {
            const float sc = __expf(mrow[v] - nm[v]);
            lrow[v] *= sc; mrow[v] = nm[v];
#pragma unroll
            for (int t = 0; t < 8; ++t) yacc[t][v] *= sc;
        }
        // exponentiate + row sum
#pragma unroll
        for (int v = 0; v < 8; ++v) {
            float acc = 0.f;
#pragma unroll
            for (int jt = 0; jt < 4; ++jt) {
                const float e = __expf(s[jt][v] - mrow[v]);
                s[jt][v] = e; acc += e;
            }
#pragma unroll
            for (int off = 1; off < 16; off <<= 1) acc += __shfl_xor(acc, off, 32);
            lrow[v] += acc;
        }
        // ---- transpose P through LDS (C-layout -> A-layout) ----
#pragma unroll
        for (int jt = 0; jt < 4; ++jt) {
#pragma unroll
            for (int v = 0; v < 8; v += 2) {
                v2bf pr = cvtpk(s[jt][v], s[jt][v + 1]);
                plds[v + 8 * h][jt * 16 + m]     = pr[0];
                plds[v + 1 + 8 * h][jt * 16 + m] = pr[1];
            }
        }
        asm volatile("s_wait_dscnt 0" ::: "memory");   // wave-local LDS fence
        v16bf pf[2];
#pragma unroll
        for (int k2 = 0; k2 < 2; ++k2) {
            v8bf lo = *(const v8bf*)&plds[m][k2 * 32 + 16 * h];
            v8bf hi = *(const v8bf*)&plds[m][k2 * 32 + 16 * h + 8];
            v16bf t;
#pragma unroll
            for (int i = 0; i < 8; ++i) { t[i] = lo[i]; t[8 + i] = hi[i]; }
            pf[k2] = t;
        }
        // ---- Y += P @ G  (16 x 128), pipelined over 16 (t,k2) steps ----
        {
            const int gbase = m * N_ + j0 + 16 * h;     // + t*16*N_ + k2*32
            v16bf gbf = *(const v16bf*)(gt + gbase);
#pragma unroll
            for (int i = 0; i < 16; ++i) {
                const int t = i >> 1, k2 = i & 1;
                v16bf gn = gbf;
                if (i < 15) {
                    const int tn = (i + 1) >> 1, kn = (i + 1) & 1;
                    gn = *(const v16bf*)(gt + gbase + tn * (16 * N_) + kn * 32);
                }
                yacc[t] = WMMA_BF16(pf[k2], gbf, yacc[t]);
                gbf = gn;
            }
        }
    }
    // ---- finalize: y / l, store (B,N,CI) bf16 ----
#pragma unroll
    for (int v = 0; v < 8; ++v) lrow[v] = 1.0f / lrow[v];
#pragma unroll
    for (int t = 0; t < 8; ++t) {
        __bf16* dst = yout + (b * N_ + n0) * CI_ + t * 16 + m;
#pragma unroll
        for (int v = 0; v < 8; ++v)
            dst[(v + 8 * h) * CI_] = f2bf(yacc[t][v] * lrow[v]);
    }
}

// ---------------------------------------------------------------------------
// K3: z = Wz @ y + bias, stored (B,C,N) fp32 (channel-major for BN reduce).
// ---------------------------------------------------------------------------
__global__ __launch_bounds__(128) void zgemm_kernel(
    const __bf16* __restrict__ y, const __bf16* __restrict__ wz,
    const float* __restrict__ wzb, float* __restrict__ z)
{
    const int wave = threadIdx.x >> 5;
    const int lane = threadIdx.x & 31;
    const int gw_id = blockIdx.x * 4 + wave;
    const int b  = gw_id / (N_ / 16);
    const int n0 = (gw_id % (N_ / 16)) * 16;
    const int m = lane & 15, h = lane >> 4;

    const __bf16* yb = y + (b * N_ + n0) * CI_;
    v16bf af[4];
#pragma unroll
    for (int k = 0; k < 4; ++k) af[k] = load_afrag(yb, CI_, m, h, k * 32);

#pragma unroll
    for (int t = 0; t < C_ / 16; ++t) {            // 16 output channel tiles
        v8f acc = {0.f,0.f,0.f,0.f,0.f,0.f,0.f,0.f};
        const int co = t * 16 + m;
        const __bf16* wrow = wz + co * CI_ + 16 * h;
        v16bf bf = *(const v16bf*)(wrow);          // software-pipelined k-loop
#pragma unroll
        for (int k = 0; k < 4; ++k) {
            v16bf bfn = bf;
            if (k < 3) bfn = *(const v16bf*)(wrow + (k + 1) * 32);
            acc = WMMA_BF16(af[k], bf, acc);
            bf = bfn;
        }
        const float bias = wzb[co];
        float* dst = z + (b * C_ + co) * N_ + n0;
#pragma unroll
        for (int v = 0; v < 8; ++v) dst[v + 8 * h] = acc[v] + bias;
    }
}

// ---------------------------------------------------------------------------
// K4: BN batch statistics per channel over (B, N).
// ---------------------------------------------------------------------------
__global__ __launch_bounds__(256) void bnstat_kernel(
    const float* __restrict__ z, float* __restrict__ mean, float* __restrict__ rstd)
{
    __shared__ float rs[256], rq[256];
    const int c = blockIdx.x;
    const int tid = threadIdx.x;
    float s = 0.f, q = 0.f;
    for (int i = tid; i < B_ * N_; i += 256) {
        const int b = i >> 12, n = i & (N_ - 1);
        const float v = z[(b * C_ + c) * N_ + n];
        s += v; q += v * v;
    }
    rs[tid] = s; rq[tid] = q;
    __syncthreads();
    for (int off = 128; off > 0; off >>= 1) {
        if (tid < off) { rs[tid] += rs[tid + off]; rq[tid] += rq[tid + off]; }
        __syncthreads();
    }
    if (tid == 0) {
        const float inv = 1.0f / (float)(B_ * N_);
        const float mu = rs[0] * inv;
        const float var = rq[0] * inv - mu * mu;
        mean[c] = mu;
        rstd[c] = rsqrtf(var + BN_EPS);
    }
}

// ---------------------------------------------------------------------------
// K5: out = gamma*(z-mean)*rstd + beta + x   (vectorized, coalesced)
// ---------------------------------------------------------------------------
__global__ __launch_bounds__(256) void bnapply_kernel(
    const float* __restrict__ z, const float* __restrict__ x,
    const float* __restrict__ gamma, const float* __restrict__ beta,
    const float* __restrict__ mean, const float* __restrict__ rstd,
    float* __restrict__ out)
{
    const int e = (blockIdx.x * 256 + threadIdx.x) * 4;   // element index
    const int c = (e >> 12) & (C_ - 1);                   // (e / N_) % C_
    const float g = gamma[c], bt = beta[c], mu = mean[c], rs = rstd[c];
    float4 zv = *(const float4*)(z + e);
    float4 xv = *(const float4*)(x + e);
    float4 o;
    o.x = g * (zv.x - mu) * rs + bt + xv.x;
    o.y = g * (zv.y - mu) * rs + bt + xv.y;
    o.z = g * (zv.z - mu) * rs + bt + xv.z;
    o.w = g * (zv.w - mu) * rs + bt + xv.w;
    *(float4*)(out + e) = o;
}

// ---------------------------------------------------------------------------
extern "C" void kernel_launch(void* const* d_in, const int* in_sizes, int n_in,
                              void* d_out, int out_size, void* d_ws, size_t ws_size,
                              hipStream_t stream) {
    const float* x   = (const float*)d_in[0];
    const float* tw  = (const float*)d_in[1];
    const float* tb  = (const float*)d_in[2];
    const float* pw  = (const float*)d_in[3];
    const float* pb  = (const float*)d_in[4];
    const float* gw  = (const float*)d_in[5];
    const float* gb  = (const float*)d_in[6];
    const float* wzw = (const float*)d_in[7];
    const float* wzb = (const float*)d_in[8];
    const float* gam = (const float*)d_in[9];
    const float* bet = (const float*)d_in[10];
    float* out = (float*)d_out;

    char* ws = (char*)d_ws;
    __bf16* theta = (__bf16*)(ws);                                   // 4 MiB
    __bf16* phiT  = (__bf16*)(ws + ((size_t)4  << 20));              // 4 MiB
    __bf16* gT    = (__bf16*)(ws + ((size_t)8  << 20));              // 4 MiB
    __bf16* yws   = (__bf16*)(ws + ((size_t)12 << 20));              // 4 MiB
    float*  z     = (float*)(ws + ((size_t)16 << 20));               // 16 MiB
    float*  mean  = (float*)(ws + ((size_t)32 << 20));
    float*  rstd  = (float*)(ws + ((size_t)32 << 20) + 4096);
    __bf16* twb   = (__bf16*)(ws + ((size_t)33 << 20));              // 64 KiB each
    __bf16* pwb   = (__bf16*)(ws + ((size_t)33 << 20) + (1 << 16));
    __bf16* gwb   = (__bf16*)(ws + ((size_t)33 << 20) + (2 << 16));
    __bf16* wzb16 = (__bf16*)(ws + ((size_t)33 << 20) + (3 << 16));

    wcvt_kernel<<<dim3((CI_ * C_) / 256), dim3(256), 0, stream>>>(
        tw, pw, gw, wzw, twb, pwb, gwb, wzb16);
    proj_kernel<<<dim3(B_ * (N_ / 16) / 4), dim3(128), 0, stream>>>(
        x, twb, tb, pwb, pb, gwb, gb, theta, phiT, gT);
    attn_kernel<<<dim3(B_ * (N_ / 16)), dim3(32), 0, stream>>>(theta, phiT, gT, yws);
    zgemm_kernel<<<dim3(B_ * (N_ / 16) / 4), dim3(128), 0, stream>>>(yws, wzb16, wzb, z);
    bnstat_kernel<<<dim3(C_), dim3(256), 0, stream>>>(z, mean, rstd);
    bnapply_kernel<<<dim3((B_ * C_ * N_ / 4) / 256), dim3(256), 0, stream>>>(
        z, x, gam, bet, mean, rstd, out);
}
